// MultiHeadAtt_28656021799090
// MI455X (gfx1250) — matrix operand order repaired and checked by reading, not compiled, not run on hardware
//
#include <hip/hip_runtime.h>
#include <hip/hip_bf16.h>

// MHA forward for MI455X (gfx1250, wave32, WMMA).
// B=2, S=2048, E=1024, H=16, D=64.

#define BATCH 2
#define SEQ   2048
#define EMB   1024
#define NH    16
#define HD    64
#define MROWS (BATCH * SEQ)   // 4096

typedef _Float16 vhalf8  __attribute__((ext_vector_type(8)));
typedef _Float16 vhalf16 __attribute__((ext_vector_type(16)));
typedef float    vfloat8 __attribute__((ext_vector_type(8)));

// ---------------------------------------------------------------------------
// D = A(16x32 f16) * B(32x16 f16) + C(16x16 f32)   -> v_wmma_f32_16x16x32_f16
// ---------------------------------------------------------------------------
__device__ __forceinline__ vfloat8 wmma_f16f32(vhalf16 a, vhalf16 b, vfloat8 c) {
  return __builtin_amdgcn_wmma_f32_16x16x32_f16(
      /*neg_a=*/false, a, /*neg_b=*/false, b,
      /*c_mod=*/(short)0, c, /*reuse_a=*/false, /*reuse_b=*/false);
}

// A-fragment (16x32, 16-bit) per ISA 7.12.2:
//   lanes 0-15 : row M=lane,    VGPR0-3 hold K=0..7,  VGPR4-7 hold K=16..23
//   lanes16-31 : row M=lane-16, VGPR0-3 hold K=8..15, VGPR4-7 hold K=24..31
// => per lane: two contiguous 16B chunks at kbase+8*half and kbase+16+8*half.
__device__ __forceinline__ vhalf16 load_a_frag(const _Float16* base, long lda,
                                               int mbase, int kbase, int lane) {
  const int row = mbase + (lane & 15);
  const int hf  = lane >> 4;
  const _Float16* p = base + (long)row * lda + kbase + hf * 8;
  union { vhalf16 v; vhalf8 h[2]; } u;
  u.h[0] = *(const vhalf8*)(p);
  u.h[1] = *(const vhalf8*)(p + 16);
  return u.v;
}

// B-fragment (32x16, 16-bit): lane = column N (mod 16); same K order as A:
// lanes 0-15 hold K=0..7,16..23, lanes 16-31 hold K=8..15,24..31. Source rows
// indexed by N (W[n, k]) -> two contiguous 16B loads per lane.
__device__ __forceinline__ vhalf16 load_b_frag(const _Float16* base, long ldb,
                                               int nbase, int kbase, int lane) {
  const int n  = nbase + (lane & 15);
  const int hf = lane >> 4;
  const _Float16* p = base + (long)n * ldb + kbase + hf * 8;
  union { vhalf16 v; vhalf8 h[2]; } u;
  u.h[0] = *(const vhalf8*)(p);
  u.h[1] = *(const vhalf8*)(p + 16);
  return u.v;
}

// ---------------------------------------------------------------------------
// fp32 -> f16 conversion (vectorized x4)
// ---------------------------------------------------------------------------
struct alignas(8) H4 { _Float16 a, b, c, d; };

__global__ void f32_to_f16_vec(const float* __restrict__ in,
                               _Float16* __restrict__ out, int n4) {
  int i = blockIdx.x * blockDim.x + threadIdx.x;
  if (i < n4) {
    const float4 f = reinterpret_cast<const float4*>(in)[i];
    H4 h = {(_Float16)f.x, (_Float16)f.y, (_Float16)f.z, (_Float16)f.w};
    reinterpret_cast<H4*>(out)[i] = h;
  }
}

// ---------------------------------------------------------------------------
// Y = A @ W^T  (A: [MROWS, EMB] f16 row-major, W: [N=EMB, K=EMB] f16 row-major)
// One wave computes a 32(M) x 64(N) tile: 2 A-fragments x 4 B-fragments =
// 8 WMMAs per K-step, software-pipelined (loads for K+32 issued before the
// WMMAs for K) and unrolled x2 so the double buffer becomes a register
// ping-pong with no rotation copies. Output address is generic:
//   m = b*SEQ + s, n = h*HD + d, off = b*sb + s*ss + h*sh + d*sd
// ---------------------------------------------------------------------------
template <bool F32OUT, bool BIAS>
__global__ void __launch_bounds__(128)
gemm32x64_nt(const _Float16* __restrict__ A, const _Float16* __restrict__ W,
             const float* __restrict__ bias,
             _Float16* __restrict__ out_h, float* __restrict__ out_f,
             long sb, long ss, long sh, long sd) {
  const int lane  = threadIdx.x & 31;
  const int wave  = threadIdx.x >> 5;
  const int mbase = blockIdx.x * 32;
  const int nbase = (blockIdx.y * 4 + wave) * 64;

  vfloat8 acc[2][4] = {};
  // prologue: fragments for k = 0
  vhalf16 a0 = load_a_frag(A, EMB, mbase,      0, lane);
  vhalf16 a1 = load_a_frag(A, EMB, mbase + 16, 0, lane);
  vhalf16 b[4];
#pragma unroll
  for (int t = 0; t < 4; ++t) b[t] = load_b_frag(W, EMB, nbase + 16 * t, 0, lane);

#pragma unroll 2
  for (int k = 0; k < EMB; k += 32) {
    const int kn = (k + 32 < EMB) ? k + 32 : k;   // next-step clamp (uniform)
    const int kp = (k + 64 < EMB) ? k + 64 : k;   // prefetch distance 2
    // L2/WGP$ prefetch two steps ahead -> global_prefetch_b8
    __builtin_prefetch(A + (long)(mbase + (lane & 15)) * EMB + kp, 0, 3);
    __builtin_prefetch(W + (long)(nbase + (lane & 15)) * EMB + kp, 0, 3);
    // ping-pong: issue loads for the NEXT step before this step's WMMAs
    vhalf16 an0 = load_a_frag(A, EMB, mbase,      kn, lane);
    vhalf16 an1 = load_a_frag(A, EMB, mbase + 16, kn, lane);
    vhalf16 bn[4];
#pragma unroll
    for (int t = 0; t < 4; ++t) bn[t] = load_b_frag(W, EMB, nbase + 16 * t, kn, lane);
#pragma unroll
    for (int t = 0; t < 4; ++t) {
      acc[0][t] = wmma_f16f32(a0, b[t], acc[0][t]);
      acc[1][t] = wmma_f16f32(a1, b[t], acc[1][t]);
    }
    a0 = an0; a1 = an1;
#pragma unroll
    for (int t = 0; t < 4; ++t) b[t] = bn[t];
  }

  // C/D layout: VGPR i, lanes 0-15 -> M=i; lanes 16-31 -> M=8+i; N = lane&15.
  const int col = lane & 15, hf = lane >> 4;
#pragma unroll
  for (int r = 0; r < 2; ++r) {
#pragma unroll
    for (int t = 0; t < 4; ++t) {
      const int n = nbase + t * 16 + col;
      const int h = n >> 6, d = n & (HD - 1);
      const float bv = BIAS ? bias[n] : 0.0f;
#pragma unroll
      for (int i = 0; i < 8; ++i) {
        const int m = mbase + r * 16 + i + 8 * hf;
        const int bb = m >> 11, s = m & (SEQ - 1);
        const long off = (long)bb * sb + (long)s * ss + (long)h * sh + (long)d * sd;
        const float v = acc[r][t][i] + bv;
        if (F32OUT) out_f[off] = v;
        else        out_h[off] = (_Float16)v;
      }
    }
  }
}

// ---------------------------------------------------------------------------
// Fused attention: per wave, one (b, h, 16-query tile); stream keys in chunks
// of 32 with online softmax. Reproduces the reference's mask bug: only keys
// with kg > qg are attended; masked entries get -1e9 (NOT -inf), so a fully
// masked row degenerates to a uniform softmax exactly like the reference.
// K-fragments are ping-pong double-buffered (loop unrolled x2) and
// V-fragments are issued before the softmax VALU block so global loads
// overlap VALU/trans work.
//   q, k : [B, H, S, D] f16      vT : [B, H, D, S] f16
//   out  : [B, S, E]    f16 (E = H*D)
// ---------------------------------------------------------------------------
__global__ void __launch_bounds__(128)
flash_attn(const _Float16* __restrict__ qh, const _Float16* __restrict__ kh,
           const _Float16* __restrict__ vT, _Float16* __restrict__ out_h) {
  __shared__ _Float16 plds_all[4][16 * 32];   // per-wave P staging (1 KB each)

  const int lane  = threadIdx.x & 31;
  const int wave  = threadIdx.x >> 5;
  const int gw    = blockIdx.x * 4 + wave;        // 0 .. B*H*(S/16)-1
  const int qtile = gw & (SEQ / 16 - 1);          // low 7 bits
  const int bh    = gw >> 7;                      // b*NH + h
  const int qb    = qtile * 16;
  const int col   = lane & 15, hf = lane >> 4;

  const _Float16* Q = qh + (long)bh * SEQ * HD;
  const _Float16* K = kh + (long)bh * SEQ * HD;
  const _Float16* V = vT + (long)bh * HD * SEQ;
  _Float16* plds = &plds_all[wave][0];

  // Q fragments for d = 0..31 and 32..63, reused across the whole key loop.
  const vhalf16 qa0 = load_a_frag(Q, HD, qb, 0,  lane);
  const vhalf16 qa1 = load_a_frag(Q, HD, qb, 32, lane);

  vfloat8 o0 = {}, o1 = {}, o2 = {}, o3 = {};  // 16 x 64 f32 accumulator
  float rm[8], rl[8];                          // running max / sum per row
#pragma unroll
  for (int i = 0; i < 8; ++i) { rm[i] = -__builtin_huge_valf(); rl[i] = 0.0f; }

  // prologue: K fragments for chunk 0  (kb[0..1]: keys+0, d 0/32; [2..3]: keys+16)
  vhalf16 kb[4];
  kb[0] = load_b_frag(K, HD, 0,  0,  lane);
  kb[1] = load_b_frag(K, HD, 0,  32, lane);
  kb[2] = load_b_frag(K, HD, 16, 0,  lane);
  kb[3] = load_b_frag(K, HD, 16, 32, lane);

#pragma unroll 2
  for (int kc = 0; kc < SEQ; kc += 32) {
    // ---- scores: S[16 x 32] = Q(16x64) . K^T, two 16x16 tiles -------------
    vfloat8 s0 = {}, s1 = {};
    s0 = wmma_f16f32(qa0, kb[0], s0);
    s0 = wmma_f16f32(qa1, kb[1], s0);
    s1 = wmma_f16f32(qa0, kb[2], s1);
    s1 = wmma_f16f32(qa1, kb[3], s1);

    // ---- issue V loads (this chunk) + K loads (next chunk) early ----------
    vhalf16 vb0 = load_b_frag(V, SEQ,  0, kc, lane);
    vhalf16 vb1 = load_b_frag(V, SEQ, 16, kc, lane);
    vhalf16 vb2 = load_b_frag(V, SEQ, 32, kc, lane);
    vhalf16 vb3 = load_b_frag(V, SEQ, 48, kc, lane);
    const int knc = (kc + 32 < SEQ) ? kc + 32 : kc;   // uniform clamp
    vhalf16 kbn[4];
    kbn[0] = load_b_frag(K, HD, knc,      0,  lane);
    kbn[1] = load_b_frag(K, HD, knc,      32, lane);
    kbn[2] = load_b_frag(K, HD, knc + 16, 0,  lane);
    kbn[3] = load_b_frag(K, HD, knc + 16, 32, lane);

    // ---- scale, mask, online softmax (overlaps the loads above) -----------
#pragma unroll
    for (int i = 0; i < 8; ++i) {
      const int qg = qb + i + 8 * hf;           // query index of this element
      const int kg = kc + col;                  // key index (tile 0)
      float v0 = (kg      > qg) ? s0[i] * 0.125f : -1e9f;   // 1/sqrt(64)
      float v1 = (kg + 16 > qg) ? s1[i] * 0.125f : -1e9f;
      float mx = fmaxf(v0, v1);
#pragma unroll
      for (int off = 1; off < 16; off <<= 1)
        mx = fmaxf(mx, __shfl_xor(mx, off, 32));            // 16-lane row max
      const float mnew = fmaxf(rm[i], mx);
      const float sf   = __expf(rm[i] - mnew);
      const float p0   = __expf(v0 - mnew);
      const float p1   = __expf(v1 - mnew);
      float ps = p0 + p1;
#pragma unroll
      for (int off = 1; off < 16; off <<= 1)
        ps += __shfl_xor(ps, off, 32);                      // 16-lane row sum
      rl[i] = rl[i] * sf + ps;
      rm[i] = mnew;
      o0[i] *= sf; o1[i] *= sf; o2[i] *= sf; o3[i] *= sf;
      // stage P (C-layout element -> row-major LDS) for A-fragment reload
      const int r = i + 8 * hf;
      plds[r * 32 + col]      = (_Float16)p0;
      plds[r * 32 + col + 16] = (_Float16)p1;
    }
    __syncthreads();

    // ---- re-layout P as an A fragment (16 x 32, K = key chunk) ------------
    union { vhalf16 v; vhalf8 h[2]; } pu;
    const _Float16* pp = plds + col * 32 + hf * 8;
    pu.h[0] = *(const vhalf8*)(pp);
    pu.h[1] = *(const vhalf8*)(pp + 16);
    __syncthreads();

    // ---- O += P(16x32) . V(32x64): vT rows are d, columns are s -----------
    o0 = wmma_f16f32(pu.v, vb0, o0);
    o1 = wmma_f16f32(pu.v, vb1, o1);
    o2 = wmma_f16f32(pu.v, vb2, o2);
    o3 = wmma_f16f32(pu.v, vb3, o3);

#pragma unroll
    for (int t = 0; t < 4; ++t) kb[t] = kbn[t];
  }

  // ---- normalize and store as [B, S, E] f16 -------------------------------
  const int b = bh >> 4, h = bh & (NH - 1);
#pragma unroll
  for (int i = 0; i < 8; ++i) {
    const float inv = 1.0f / rl[i];
    const int s = qb + i + 8 * hf;
    const long rowoff = ((long)b * SEQ + s) * EMB + h * HD;
    out_h[rowoff +  0 + col] = (_Float16)(o0[i] * inv);
    out_h[rowoff + 16 + col] = (_Float16)(o1[i] * inv);
    out_h[rowoff + 32 + col] = (_Float16)(o2[i] * inv);
    out_h[rowoff + 48 + col] = (_Float16)(o3[i] * inv);
  }
}

// ---------------------------------------------------------------------------
// Host launch
// ---------------------------------------------------------------------------
extern "C" void kernel_launch(void* const* d_in, const int* in_sizes, int n_in,
                              void* d_out, int out_size, void* d_ws, size_t ws_size,
                              hipStream_t stream) {
  const float* x  = (const float*)d_in[0];
  const float* wq = (const float*)d_in[1];
  const float* wk = (const float*)d_in[2];
  const float* wv = (const float*)d_in[3];
  const float* wo = (const float*)d_in[4];
  const float* bo = (const float*)d_in[5];
  float* out = (float*)d_out;

  char* ws = (char*)d_ws;   // 48 MiB layout
  _Float16* xh  = (_Float16*)(ws + ( 0L << 20));  // x    f16   8 MiB
  _Float16* wqh = (_Float16*)(ws + ( 8L << 20));  // wq   f16   2 MiB
  _Float16* wkh = (_Float16*)(ws + (10L << 20));
  _Float16* wvh = (_Float16*)(ws + (12L << 20));
  _Float16* woh = (_Float16*)(ws + (14L << 20));
  _Float16* qh  = (_Float16*)(ws + (16L << 20));  // q [B,H,S,D]   8 MiB
  _Float16* kh  = (_Float16*)(ws + (24L << 20));  // k [B,H,S,D]   8 MiB
  _Float16* vTh = (_Float16*)(ws + (32L << 20));  // v [B,H,D,S]   8 MiB
  _Float16* ah  = (_Float16*)(ws + (40L << 20));  // attn out [B,S,E] 8 MiB

  // fp32 -> f16 staging
  {
    const int n4 = MROWS * EMB / 4;
    f32_to_f16_vec<<<(n4 + 255) / 256, 256, 0, stream>>>(x, xh, n4);
    const int w4 = EMB * EMB / 4;
    f32_to_f16_vec<<<(w4 + 255) / 256, 256, 0, stream>>>(wq, wqh, w4);
    f32_to_f16_vec<<<(w4 + 255) / 256, 256, 0, stream>>>(wk, wkh, w4);
    f32_to_f16_vec<<<(w4 + 255) / 256, 256, 0, stream>>>(wv, wvh, w4);
    f32_to_f16_vec<<<(w4 + 255) / 256, 256, 0, stream>>>(wo, woh, w4);
  }

  const dim3 gg(MROWS / 32, 4), gb(128);
  // q, k -> [B, H, S, D]
  gemm32x64_nt<false, false><<<gg, gb, 0, stream>>>(xh, wqh, nullptr, qh, nullptr,
      (long)NH * SEQ * HD, (long)HD, (long)SEQ * HD, 1L);
  gemm32x64_nt<false, false><<<gg, gb, 0, stream>>>(xh, wkh, nullptr, kh, nullptr,
      (long)NH * SEQ * HD, (long)HD, (long)SEQ * HD, 1L);
  // v -> [B, H, D, S] (pre-transposed so PV B-fragments are contiguous)
  gemm32x64_nt<false, false><<<gg, gb, 0, stream>>>(xh, wvh, nullptr, vTh, nullptr,
      (long)NH * HD * SEQ, 1L, (long)HD * SEQ, (long)SEQ);

  // fused attention: B*H*(S/16) = 4096 waves, 4 waves per block
  flash_attn<<<(BATCH * NH * (SEQ / 16)) / 4, 128, 0, stream>>>(qh, kh, vTh, ah);

  // out = attn @ wo^T + bo  -> fp32 [B, S, E]
  gemm32x64_nt<true, true><<<gg, gb, 0, stream>>>(ah, woh, bo, nullptr, out,
      (long)SEQ * EMB, (long)EMB, (long)HD, 1L);
}
